// L2OOptimizer_80504866996656
// MI455X (gfx1250) — matrix operand order repaired and checked by reading, not compiled, not run on hardware
//
#include <hip/hip_runtime.h>
#include <hip/hip_bf16.h>

typedef __attribute__((ext_vector_type(16))) __bf16 v16bf;
typedef __attribute__((ext_vector_type(8)))  __bf16 v8bf;
typedef __attribute__((ext_vector_type(8)))  float  v8f;
typedef __attribute__((ext_vector_type(4)))  float  f32x4;

#define ROWS 128      // rows (b,n pairs) per workgroup tile
#define KC   96       // combined K: 64 (hidden) + 32 (t)
#define NC   64       // output columns (STATE)
#define XP   104      // LDS row pitch, input tile (bf16 units); 208B -> 16B aligned
#define WP   104      // LDS row pitch, [A;Bm] weight tile (bf16 units)
#define HP   40       // LDS row pitch, h1 tile (bf16 units); 80B -> 16B aligned
#define W2P  40       // LDS row pitch, W2^T tile (bf16 units)

// Build a 16-element bf16 fragment from two 16-byte-aligned LDS chunks.
__device__ __forceinline__ v16bf ld_v16(const __bf16* p) {
    v8bf lo = *(const v8bf*)(p);
    v8bf hi = *(const v8bf*)(p + 8);
    v16bf r;
#pragma unroll
    for (int i = 0; i < 8; i++) { r[i] = lo[i]; r[i + 8] = hi[i]; }
    return r;
}

// CDNA5 has a native V_TANH_F32 transcendental (TRANS pipe, co-executes with
// WMMA). Prefer the builtin; fall back to inline asm, which both toolchains'
// assemblers accept for gfx1250.
__device__ __forceinline__ float fast_tanh(float x) {
#if defined(__has_builtin)
#if __has_builtin(__builtin_amdgcn_tanhf)
    return __builtin_amdgcn_tanhf(x);
#else
    float r;
    asm volatile("v_tanh_f32 %0, %1" : "=v"(r) : "v"(x));
    return r;
#endif
#else
    float r;
    asm volatile("v_tanh_f32 %0, %1" : "=v"(r) : "v"(x));
    return r;
#endif
}

__global__ __launch_bounds__(256)
void l2o_fused_kernel(const float* __restrict__ x,
                      const float* __restrict__ grad,
                      const float* __restrict__ hidden,
                      const float* __restrict__ W1,
                      const float* __restrict__ b1,
                      const float* __restrict__ W2,
                      const float* __restrict__ b2,
                      const float* __restrict__ A,
                      const float* __restrict__ Bm,
                      const float* __restrict__ C,
                      float* __restrict__ x_out,
                      float* __restrict__ h_out,
                      float* __restrict__ delta_out)
{
    __shared__ __bf16 Xt[ROWS * XP];    // [hidden | t] tile, bf16, row-major (M x 96)
    __shared__ __bf16 Wt[NC * WP];      // [A ; Bm]^T : Wt[n][k], bf16 (64 x 96)
    __shared__ __bf16 H1[ROWS * HP];    // relu layer-1 activations, bf16 (M x 32)
    __shared__ __bf16 W2T[32 * W2P];    // W2^T : W2T[n][k], bf16 (32 x 32)
    __shared__ float  lC[64];

    const int  tid  = threadIdx.x;
    const int  lane = tid & 31;
    const int  wave = tid >> 5;
    const int  ksel = lane >> 4;        // 16-bit fragment half select per ISA layout
    const long long row0 = (long long)blockIdx.x * ROWS;

    // ================= Phase A: stage weights + inputs into LDS =================
    // Combined GEMM weights [A ; Bm] as bf16, column(n)-major for B fragments
    for (int e = tid; e < NC * KC; e += 256) {
        int n = e / KC;
        int k = e - n * KC;
        float v = (k < 64) ? A[k * 64 + n] : Bm[(k - 64) * 64 + n];
        Wt[n * WP + k] = (__bf16)v;
    }
    // W2^T as bf16 for the MLP layer-2 WMMA
    for (int e = tid; e < 32 * 32; e += 256) {
        int n = e >> 5;
        int k = e & 31;
        W2T[n * W2P + k] = (__bf16)W2[k * 32 + n];
    }
    if (tid < 64) lC[tid] = C[tid];

    // hidden tile (128 x 64 f32) -> bf16 LDS; streaming (NT) coalesced loads
    const f32x4* h4 = (const f32x4*)(hidden + row0 * 64);
#pragma unroll
    for (int it = 0; it < 8; it++) {
        int idx = tid + it * 256;       // 0..2047 float4s (128 rows * 16/row)
        f32x4 v = __builtin_nontemporal_load(&h4[idx]);
        int r = idx >> 4;
        int c = (idx & 15) << 2;
        __bf16* d = &Xt[r * XP + c];
        d[0] = (__bf16)v.x; d[1] = (__bf16)v.y; d[2] = (__bf16)v.z; d[3] = (__bf16)v.w;
    }

    // MLP layer 1 on VALU (tiny: K=2): 2 threads per row, 16 outputs each.
    // Wave w writes rows 16w..16w+15 == the rows its own WMMA-2 A-frag reads,
    // so same-wave LDS ordering suffices (no extra barrier for H1).
    {
        int r  = tid >> 1;
        int j0 = (tid & 1) << 4;
        float g = grad[row0 + r];
        float s = (g > 0.f) ? 1.f : ((g < 0.f) ? -1.f : 0.f);
        float l = __logf(fabsf(g) + 1e-16f);
#pragma unroll
        for (int j = 0; j < 16; j++) {
            float v = s * W1[j0 + j] + l * W1[32 + j0 + j] + b1[j0 + j];
            H1[r * HP + j0 + j] = (__bf16)((v > 0.f) ? v : 0.f);
        }
    }
    __syncthreads();   // W2T / Wt / Xt visible to all waves

    // ============ Phase B: t = relu(...) @ W2 + b2 via WMMA (2 per wave) ========
    {
        const int rbase = wave * 16 + ksel * 8;
        v16bf ah = ld_v16(&H1[(wave * 16 + (lane & 15)) * HP + (ksel << 4)]);
#pragma unroll
        for (int nt = 0; nt < 2; nt++) {
            const int ncol = nt * 16 + (lane & 15);
            v16bf bw = ld_v16(&W2T[ncol * W2P + (ksel << 4)]);
            v8f acc = {0.f, 0.f, 0.f, 0.f, 0.f, 0.f, 0.f, 0.f};
            acc = __builtin_amdgcn_wmma_f32_16x16x32_bf16(
                      false, ah, false, bw, (short)0, acc, false, false);
            float b2v = b2[ncol];
#pragma unroll
            for (int r = 0; r < 8; r++)
                Xt[(rbase + r) * XP + 64 + ncol] = (__bf16)(acc[r] + b2v);
        }
    }
    __syncthreads();   // t columns written cross-lane within the wave

    // ===== Phase C: h_new = tanh([hidden|t] @ [A;Bm]) — 12 WMMAs per wave =====
    const int mloc = wave * 16 + (lane & 15);
    v16bf afr[3];
#pragma unroll
    for (int kt = 0; kt < 3; kt++)
        afr[kt] = ld_v16(&Xt[mloc * XP + kt * 32 + (ksel << 4)]);

    float dpart[8];
#pragma unroll
    for (int r = 0; r < 8; r++) dpart[r] = 0.f;

    const long long mbase = row0 + wave * 16 + ksel * 8;  // C/D rows for this half

#pragma unroll
    for (int nt = 0; nt < 4; nt++) {
        v8f acc = {0.f, 0.f, 0.f, 0.f, 0.f, 0.f, 0.f, 0.f};
        const int ncol = nt * 16 + (lane & 15);
#pragma unroll
        for (int kt = 0; kt < 3; kt++) {
            v16bf bfr = ld_v16(&Wt[ncol * WP + kt * 32 + (ksel << 4)]);
            acc = __builtin_amdgcn_wmma_f32_16x16x32_bf16(
                      false, afr[kt], false, bfr, (short)0, acc, false, false);
        }
        float cv = lC[ncol];
#pragma unroll
        for (int r = 0; r < 8; r++) {
            float hv = fast_tanh(acc[r]);
            __builtin_nontemporal_store(hv, &h_out[(mbase + r) * 64 + ncol]);
            dpart[r] += hv * cv;
        }
    }

    // delta = h_new . C : butterfly reduction across the 16-lane N-group
#pragma unroll
    for (int r = 0; r < 8; r++) {
        float d = dpart[r];
        d += __shfl_xor(d, 1, 32);
        d += __shfl_xor(d, 2, 32);
        d += __shfl_xor(d, 4, 32);
        d += __shfl_xor(d, 8, 32);
        dpart[r] = d;
    }
    if ((lane & 15) == 0) {
#pragma unroll
        for (int r = 0; r < 8; r++) {
            long long m = mbase + r;
            float d = dpart[r];
            delta_out[m] = d;
            x_out[m]     = x[m] + d;
        }
    }
}

extern "C" void kernel_launch(void* const* d_in, const int* in_sizes, int n_in,
                              void* d_out, int out_size, void* d_ws, size_t ws_size,
                              hipStream_t stream) {
    const float* x      = (const float*)d_in[0];
    const float* grad   = (const float*)d_in[1];
    const float* hidden = (const float*)d_in[2];
    const float* W1     = (const float*)d_in[3];
    const float* b1     = (const float*)d_in[4];
    const float* W2     = (const float*)d_in[5];
    const float* b2     = (const float*)d_in[6];
    const float* A      = (const float*)d_in[7];
    const float* Bm     = (const float*)d_in[8];
    const float* C      = (const float*)d_in[9];

    const long long Mtot = (long long)in_sizes[0];   // B*N = 1,600,000 (multiple of 128)

    float* x_out     = (float*)d_out;                // (B,N)
    float* h_out     = x_out + Mtot;                 // (B,N,64)
    float* delta_out = h_out + Mtot * 64;            // (B,N)

    const int grid = (int)((Mtot + ROWS - 1) / ROWS);
    l2o_fused_kernel<<<grid, 256, 0, stream>>>(x, grad, hidden, W1, b1, W2, b2,
                                               A, Bm, C, x_out, h_out, delta_out);
}
